// Model_24146306138598
// MI455X (gfx1250) — matrix operand order repaired and checked by reading, not compiled, not run on hardware
//
#include <hip/hip_runtime.h>
#include <math.h>
#include <stdint.h>

// ---------------------------------------------------------------------------
// Reformer (LSH attention) forward for MI455X / gfx1250.
// Dense math: v_wmma_f32_16x16x32_bf16 (bf16 in, f32 accum).
// GEMM operand staging: Tensor Data Mover (tensor_load_to_lds) double-buffered
// into LDS, synchronized with s_wait_tensorcnt + workgroup barriers.
// ---------------------------------------------------------------------------

typedef __attribute__((ext_vector_type(16))) __bf16 v16bf;
typedef __attribute__((ext_vector_type(8)))  float  v8f;

#if defined(__gfx1250__) && __has_builtin(__builtin_amdgcn_tensor_load_to_lds) && \
    __has_builtin(__builtin_amdgcn_s_wait_tensorcnt)
#define HAVE_TDM 1
#else
#define HAVE_TDM 0
#endif

namespace {
constexpr int Bsz  = 4;
constexpr int SEQ  = 3584;
constexpr int PRED = 512;
constexpr int CIN  = 7;
constexpr int COUT = 7;
constexpr int L    = SEQ + PRED;      // 4096
constexpr int D    = 512;
constexpr int H    = 8;
constexpr int DH   = 64;
constexpr int FF   = 2048;
constexpr int EL   = 2;
constexpr int BS   = 64;
constexpr int NHASH = 4;
constexpr int NB   = L / BS;          // 64
constexpr int BH   = Bsz * H;         // 32
constexpr int M    = Bsz * L;         // 16384
constexpr int SL   = NHASH * L;       // 16384 sorted length per bh
constexpr int NCH  = SL / BS;         // 256 chunks per bh
constexpr float SELF_VAL = -5e4f;
}

// ---------------------------------------------------------------------------
// WMMA fragment loaders from LDS (CDNA5 16-bit layouts, wave32).
// A (16x32, MxK): lanes 0-15 -> M=lane, lanes 16-31 -> M=lane-16 with K+8;
//   VGPR v: K = (v>=4?16:0) + 2*(v&3) (+8 for hi lanes), pair (K, K+1).
// B (32x16, KxN), stored transposed in LDS as Bt[n][k]:
//   lane: N = lane&15, K-offset = (lane&16)?8:0; half h -> K = off + (h<8 ? h : h+8).
// C/D (16x16 f32): VGPR r -> M = r + ((lane&16)?8:0), N = lane&15.
// ---------------------------------------------------------------------------
__device__ inline v16bf frag_a_ld(const __bf16* base, int ld, int mi, int k0, int lane) {
  v16bf a;
  int m  = mi + (lane & 15);
  int kl = (lane & 16) ? 8 : 0;
  const __bf16* p = base + m * ld + k0 + kl;
#pragma unroll
  for (int v = 0; v < 8; ++v) {
    int kb = ((v & 4) ? 16 : 0) + 2 * (v & 3);
    a[2 * v]     = p[kb];
    a[2 * v + 1] = p[kb + 1];
  }
  return a;
}

__device__ inline v16bf frag_bt_ld(const __bf16* base, int ld, int ni, int k0, int lane) {
  v16bf b;
  int n = ni + (lane & 15);
  const __bf16* p = base + n * ld + k0 + ((lane & 16) ? 8 : 0);
#pragma unroll
  for (int h = 0; h < 16; ++h) b[h] = p[(h < 8) ? h : h + 8];
  return b;
}

__device__ inline float gelu_exact(float x) {
  return 0.5f * x * (1.0f + erff(x * 0.70710678118654752f));
}

#if HAVE_TDM
typedef __attribute__((ext_vector_type(4))) unsigned int u32x4;
typedef __attribute__((ext_vector_type(8))) int i32x8;
typedef __attribute__((ext_vector_type(4))) int i32x4;

// Issue a TDM 2-D tile load: bf16 tensor [R][K] row-major (stride K elems),
// tile = [tile_rows][32] at (row0, k0) -> LDS (contiguous [tile_rows][32]).
__device__ inline void tdm_tile_load(const __bf16* base, int R, int K,
                                     int row0, int k0, int tile_rows,
                                     unsigned lds_off) {
  unsigned long long ga =
      (unsigned long long)(uintptr_t)(base + (size_t)row0 * K + k0);
  u32x4 g0;
  g0[0] = 1u;                                   // count=1, user mode
  g0[1] = lds_off;                              // LDS byte address
  g0[2] = (unsigned)(ga & 0xFFFFFFFFu);         // global_addr[31:0]
  g0[3] = (unsigned)((ga >> 32) & 0x01FFFFFFu)  // global_addr[56:32]
        | (2u << 30);                           // type = 2 ("image")
  i32x8 g1;
  g1[0] = (int)(1u << 16);                      // data_size = 1 -> 2 bytes
  g1[1] = (int)(((unsigned)K & 0xFFFFu) << 16); // tensor_dim0[15:0] @ bit48
  g1[2] = (int)((((unsigned)K >> 16) & 0xFFFFu)
        | (((unsigned)R & 0xFFFFu) << 16));     // td0 hi | tensor_dim1 lo
  g1[3] = (int)((((unsigned)R >> 16) & 0xFFFFu)
        | (32u << 16));                         // td1 hi | tile_dim0 = 32
  g1[4] = (int)((unsigned)tile_rows & 0xFFFFu); // tile_dim1 | tile_dim2=0
  g1[5] = (int)(unsigned)K;                     // tensor_dim0_stride low32
  g1[6] = 0;                                    // stride hi16 | dim1_stride lo
  g1[7] = 0;
  i32x4 z4 = {0, 0, 0, 0};
  i32x8 z8 = {0, 0, 0, 0, 0, 0, 0, 0};
  __builtin_amdgcn_tensor_load_to_lds(g0, g1, z4, z4, z8, 0);
}
#endif

// ---------------------------------------------------------------------------
// f32 -> bf16 conversion (activations), and transposed weight conversion
// Wt[n][k] = (bf16)W[k][n] so the GEMM B-tile is a plain 2-D strided copy.
// ---------------------------------------------------------------------------
__global__ __launch_bounds__(256) void tobf16_kernel(
    const float* __restrict__ X, __bf16* __restrict__ Y, size_t n) {
  size_t i = (size_t)blockIdx.x * 256 + threadIdx.x;
  if (i < n) Y[i] = (__bf16)X[i];
}

__global__ __launch_bounds__(256) void wtrans_kernel(
    const float* __restrict__ W, __bf16* __restrict__ Wt, int K, int N) {
  int k = blockIdx.x * 32 + (threadIdx.x & 31);
  int n = blockIdx.y * 8 + (threadIdx.x >> 5);
  if (k < K && n < N) Wt[(size_t)n * K + k] = (__bf16)W[(size_t)k * N + n];
}

// ---------------------------------------------------------------------------
// Embedding: circular Conv1d(k=3) token emb + 5 temporal lookups + sinusoidal PE
// ---------------------------------------------------------------------------
__global__ __launch_bounds__(256) void embed_kernel(
    const float* __restrict__ x_enc, const float* __restrict__ x_dec,
    const int* __restrict__ xm_enc, const int* __restrict__ xm_dec,
    const float* __restrict__ conv_w,
    const float* __restrict__ month_e, const float* __restrict__ day_e,
    const float* __restrict__ wday_e, const float* __restrict__ hour_e,
    const float* __restrict__ min_e, float* __restrict__ out) {
  int bt = blockIdx.x;
  int b = bt / L, t = bt % L;
  int tid = threadIdx.x;
  __shared__ float xw[3][CIN];
  __shared__ int marks[5];
  if (tid < 3 * CIN) {
    int k = tid / CIN, c = tid % CIN;
    int tt = (t - 1 + k + L) % L;
    float xv = (tt < SEQ) ? x_enc[((size_t)b * SEQ + tt) * CIN + c]
                          : x_dec[((size_t)b * PRED + (tt - SEQ)) * CIN + c];
    xw[k][c] = xv;
  }
  if (tid < 5) {
    marks[tid] = (t < SEQ) ? xm_enc[((size_t)b * SEQ + t) * 5 + tid]
                           : xm_dec[((size_t)b * PRED + (t - SEQ)) * 5 + tid];
  }
  __syncthreads();
  for (int d = tid; d < D; d += 256) {
    float acc = 0.f;
#pragma unroll
    for (int k = 0; k < 3; ++k)
#pragma unroll
      for (int c = 0; c < CIN; ++c)
        acc += xw[k][c] * conv_w[(k * CIN + c) * D + d];
    acc += month_e[(size_t)marks[0] * D + d];
    acc += day_e[(size_t)marks[1] * D + d];
    acc += wday_e[(size_t)marks[2] * D + d];
    acc += hour_e[(size_t)marks[3] * D + d];
    acc += min_e[(size_t)marks[4] * D + d];
    int i2 = d >> 1;
    float div = __expf((float)(2 * i2) * (-9.210340371976184f / (float)D));
    float ang = (float)t * div;
    acc += (d & 1) ? __cosf(ang) : __sinf(ang);
    out[((size_t)b * L + t) * D + d] = acc;
  }
}

// ---------------------------------------------------------------------------
// GEMM: C[M,N] = A[M,K] @ B[K,N] (+bias, +gelu, +residual).
// A: bf16 [M][K]; Bt: bf16 [N][K] (pre-transposed). Tile 128x64x32.
// 8 waves; wave w -> row tile w, 4 col tiles (4 WMMA / wave / K-step).
// Staging: TDM tensor_load_to_lds double-buffered (fallback: manual copy).
// ---------------------------------------------------------------------------
__global__ __launch_bounds__(256) void gemm_kernel(
    const __bf16* __restrict__ A, const __bf16* __restrict__ Bt,
    const float* __restrict__ bias, const float* __restrict__ res,
    float* __restrict__ C, int Mrows, int N, int K, int act) {
  __shared__ __bf16 sA[2][128 * 32];   // [m][k], ld=32
  __shared__ __bf16 sB[2][64 * 32];    // [n][k], ld=32
  int tid = threadIdx.x;
  int lane = tid & 31, wave = tid >> 5;
  int bm = blockIdx.x * 128;
  int bn = blockIdx.y * 64;
  int mi = wave * 16;
  v8f acc[4] = {};

#if HAVE_TDM
  unsigned ldsA[2] = {(unsigned)(uintptr_t)&sA[0][0], (unsigned)(uintptr_t)&sA[1][0]};
  unsigned ldsB[2] = {(unsigned)(uintptr_t)&sB[0][0], (unsigned)(uintptr_t)&sB[1][0]};
  if (wave == 0) {
    tdm_tile_load(A, Mrows, K, bm, 0, 128, ldsA[0]);
    tdm_tile_load(Bt, N, K, bn, 0, 64, ldsB[0]);
  }
  int buf = 0;
  for (int k0 = 0; k0 < K; k0 += 32, buf ^= 1) {
    int kn = k0 + 32;
    bool more = kn < K;
    if (wave == 0) {
      if (more) {
        tdm_tile_load(A, Mrows, K, bm, kn, 128, ldsA[buf ^ 1]);
        tdm_tile_load(Bt, N, K, bn, kn, 64, ldsB[buf ^ 1]);
        __builtin_amdgcn_s_wait_tensorcnt(2);
      } else {
        __builtin_amdgcn_s_wait_tensorcnt(0);
      }
    }
    __syncthreads();
    v16bf af = frag_a_ld(sA[buf], 32, mi, 0, lane);
#pragma unroll
    for (int jt = 0; jt < 4; ++jt) {
      v16bf bfv = frag_bt_ld(sB[buf], 32, jt * 16, 0, lane);
      acc[jt] = __builtin_amdgcn_wmma_f32_16x16x32_bf16(
          false, af, false, bfv, (short)0, acc[jt], false, false);
    }
    __syncthreads();
  }
#else
  for (int k0 = 0; k0 < K; k0 += 32) {
    for (int idx = tid; idx < 128 * 32; idx += 256) {
      int m = idx >> 5, k = idx & 31;
      sA[0][idx] = A[(size_t)(bm + m) * K + k0 + k];
    }
    for (int idx = tid; idx < 64 * 32; idx += 256) {
      int n = idx >> 5, k = idx & 31;
      sB[0][idx] = Bt[(size_t)(bn + n) * K + k0 + k];
    }
    __syncthreads();
    v16bf af = frag_a_ld(sA[0], 32, mi, 0, lane);
#pragma unroll
    for (int jt = 0; jt < 4; ++jt) {
      v16bf bfv = frag_bt_ld(sB[0], 32, jt * 16, 0, lane);
      acc[jt] = __builtin_amdgcn_wmma_f32_16x16x32_bf16(
          false, af, false, bfv, (short)0, acc[jt], false, false);
    }
    __syncthreads();
  }
#endif

  int mrow = (lane & 16) ? 8 : 0;
  int nn = lane & 15;
#pragma unroll
  for (int jt = 0; jt < 4; ++jt) {
#pragma unroll
    for (int r = 0; r < 8; ++r) {
      int m = bm + mi + mrow + r;
      int col = bn + jt * 16 + nn;
      float v = acc[jt][r];
      if (bias) v += bias[col];
      if (act)  v = gelu_exact(v);
      if (res)  v += res[(size_t)m * N + col];
      C[(size_t)m * N + col] = v;
    }
  }
}

// ---------------------------------------------------------------------------
// LSH hashing: per (bh,t) compute 4x32 rotations, argmax over [rv, -rv].
// ---------------------------------------------------------------------------
__global__ __launch_bounds__(128) void hash_kernel(
    const float* __restrict__ QK, const float* __restrict__ rot,
    int* __restrict__ buckets) {
  int gid = blockIdx.x;            // bh * L + t
  int bh = gid / L, t = gid % L;
  int b = bh >> 3, hd = bh & 7;
  int tid = threadIdx.x;
  int h = tid >> 5, i = tid & 31;
  __shared__ float qrow[DH];
  __shared__ float dv[128];
  if (tid < DH) qrow[tid] = QK[((size_t)b * L + t) * D + hd * DH + tid];
  __syncthreads();
  float d = 0.f;
#pragma unroll 8
  for (int f = 0; f < DH; ++f) d += qrow[f] * rot[(size_t)f * (NHASH * (NB / 2)) + h * (NB / 2) + i];
  dv[tid] = d;
  __syncthreads();
  if (i == 0) {
    float best = -3.0e38f; int bi = 0;
    for (int j = 0; j < 32; ++j) { float x = dv[h * 32 + j]; if (x > best) { best = x; bi = j; } }
    for (int j = 0; j < 32; ++j) { float x = -dv[h * 32 + j]; if (x > best) { best = x; bi = 32 + j; } }
    buckets[((size_t)bh * NHASH + h) * L + t] = bi;
  }
}

// ---------------------------------------------------------------------------
// Stable counting sort per (bh, hash): 4096 keys into 64 buckets, tie by pos.
// st[bh][h*L+pos] = t ; undo[bh][h*L+t] = h*L+pos (argsort on unique keys)
// ---------------------------------------------------------------------------
__global__ __launch_bounds__(128) void sort_kernel(
    const int* __restrict__ buckets, int* __restrict__ st, int* __restrict__ undo) {
  int bhh = blockIdx.x;            // bh*NHASH + h
  int bh = bhh >> 2, h = bhh & 3;
  __shared__ unsigned char lb[L];
  __shared__ int hist[NB];
  __shared__ int off[NB];
  int tid = threadIdx.x;
  if (tid < NB) hist[tid] = 0;
  __syncthreads();
  for (int t = tid; t < L; t += 128) {
    int bk = buckets[(size_t)bhh * L + t];
    lb[t] = (unsigned char)bk;
    atomicAdd(&hist[bk], 1);
  }
  __syncthreads();
  if (tid == 0) {
    int acc = 0;
    for (int k = 0; k < NB; ++k) { off[k] = acc; acc += hist[k]; }
  }
  __syncthreads();
  if (tid < NB) {
    int bucket = tid;
    int pos = off[bucket];
    size_t base = (size_t)bh * SL + (size_t)h * L;
    for (int t = 0; t < L; ++t) {
      if (lb[t] == bucket) {
        st[base + pos] = t;
        undo[base + t] = h * L + pos;
        ++pos;
      }
    }
  }
}

// ---------------------------------------------------------------------------
// Bucketed attention chunk: one block per (bh, chunk c).
// dots = bq(64x64) @ bk^T(64x128) via WMMA; mask; softmax; bo = p @ bvv.
// ---------------------------------------------------------------------------
__global__ __launch_bounds__(256) void attn_kernel(
    const float* __restrict__ QK, const float* __restrict__ V,
    const int* __restrict__ st, float* __restrict__ so, float* __restrict__ slog) {
  int c = blockIdx.x;
  int bh = blockIdx.y;
  int b = bh >> 3, hd = bh & 7;
  int tid = threadIdx.x;
  int lane = tid & 31, wave = tid >> 5;

  __shared__ float  s_dots[64 * 128];               // 32KB; later bf16 p packed per-row
  __shared__ __bf16 s_kq[64 * 64 + 128 * 64];       // 24KB: bq | bk ; reused as vT[64][128]
  __shared__ float  s_inv[128];
  __shared__ int    s_tk[128];

  if (tid < 128) {
    int r = tid;
    int p = (r < 64) ? (c * BS + r) : (((c + NCH - 1) % NCH) * BS + (r - 64));
    s_tk[r] = st[(size_t)bh * SL + p];
  }
  __syncthreads();

  for (int idx = tid; idx < 128 * 64; idx += 256) {
    int r = idx >> 6, e = idx & 63;
    float x = QK[((size_t)b * L + s_tk[r]) * D + hd * DH + e];
    if (r < 64) s_kq[r * 64 + e] = (__bf16)x;
    s_kq[64 * 64 + idx] = (__bf16)x;
  }
  if (tid < 128) {
    int r = tid;
    const float* row = QK + ((size_t)b * L + s_tk[r]) * D + hd * DH;
    float s = 0.f;
#pragma unroll 8
    for (int e = 0; e < 64; ++e) { float x = row[e]; s += x * x; }
    s_inv[r] = 1.0f / fmaxf(sqrtf(s), 1e-12f);
  }
  __syncthreads();
  for (int idx = tid; idx < 128 * 64; idx += 256) {
    int r = idx >> 6;
    float x = (float)s_kq[64 * 64 + idx] * s_inv[r];
    s_kq[64 * 64 + idx] = (__bf16)x;
  }
  __syncthreads();

  {
    const __bf16* bq = s_kq;
    const __bf16* bk = s_kq + 64 * 64;
    int mi = (wave >> 1) * 16;
    int nb0 = (wave & 1) * 64;
    int mrow = (lane & 16) ? 8 : 0;
    int nn = lane & 15;
#pragma unroll
    for (int jt = 0; jt < 4; ++jt) {
      int ni = nb0 + jt * 16;
      v8f acc = {};
#pragma unroll
      for (int k0 = 0; k0 < 64; k0 += 32) {
        v16bf af = frag_a_ld(bq, 64, mi, k0, lane);
        v16bf bf = frag_bt_ld(bk, 64, ni, k0, lane);
        acc = __builtin_amdgcn_wmma_f32_16x16x32_bf16(false, af, false, bf, (short)0, acc, false, false);
      }
#pragma unroll
      for (int r = 0; r < 8; ++r) {
        int i = mi + mrow + r, j = ni + nn;
        float v = acc[r] * 0.125f;                 // dh^-0.5
        if (s_tk[i] == s_tk[j]) v = SELF_VAL;      // self-mask
        s_dots[i * 128 + j] = v;
      }
    }
  }
  __syncthreads();

  if (tid < 64) {
    int i = tid;
    float mx = -3.0e38f;
    for (int j = 0; j < 128; ++j) mx = fmaxf(mx, s_dots[i * 128 + j]);
    float s = 0.f;
    for (int j = 0; j < 128; ++j) s += __expf(s_dots[i * 128 + j] - mx);
    float lse = mx + __logf(s);
    slog[(size_t)bh * SL + c * BS + i] = lse;
    __bf16* prow = reinterpret_cast<__bf16*>(s_dots);
    for (int j = 0; j < 128; ++j) {
      float pv = __expf(s_dots[i * 128 + j] - lse);
      prow[i * 256 + j] = (__bf16)pv;              // in-place pack (safe: 2j<=4j)
    }
  } else {
    for (int idx = tid - 64; idx < 128 * 64; idx += 192) {
      int r = idx >> 6, e = idx & 63;
      s_kq[e * 128 + r] = (__bf16)V[((size_t)b * L + s_tk[r]) * D + hd * DH + e];
    }
  }
  __syncthreads();

  {
    const __bf16* pA = reinterpret_cast<const __bf16*>(s_dots); // ld=256 halves
    const __bf16* vT = s_kq;                                     // vT[e][j], ld=128
    int mi = (wave >> 1) * 16;
    int mrow = (lane & 16) ? 8 : 0;
    int nn = lane & 15;
#pragma unroll
    for (int jt = 0; jt < 2; ++jt) {
      int ni = ((wave & 1) * 2 + jt) * 16;
      v8f acc = {};
#pragma unroll
      for (int k0 = 0; k0 < 128; k0 += 32) {
        v16bf af = frag_a_ld(pA, 256, mi, k0, lane);
        v16bf bf = frag_bt_ld(vT, 128, ni, k0, lane);
        acc = __builtin_amdgcn_wmma_f32_16x16x32_bf16(false, af, false, bf, (short)0, acc, false, false);
      }
#pragma unroll
      for (int r = 0; r < 8; ++r) {
        int i = mi + mrow + r, e = ni + nn;
        so[((size_t)bh * SL + c * BS + i) * DH + e] = acc[r];
      }
    }
  }
}

// ---------------------------------------------------------------------------
// Un-sort + softmax-combine across the 4 hash rounds; scatter into (B,L,D).
// ---------------------------------------------------------------------------
__global__ __launch_bounds__(256) void combine_kernel(
    const float* __restrict__ so, const float* __restrict__ slog,
    const int* __restrict__ undo, float* __restrict__ attn) {
  int bh = blockIdx.y;
  int t = blockIdx.x * 4 + (threadIdx.x >> 6);
  int e = threadIdx.x & 63;
  size_t base = (size_t)bh * SL;
  int k0 = undo[base + 0 * L + t];
  int k1 = undo[base + 1 * L + t];
  int k2 = undo[base + 2 * L + t];
  int k3 = undo[base + 3 * L + t];
  float l0 = slog[base + k0], l1 = slog[base + k1];
  float l2 = slog[base + k2], l3 = slog[base + k3];
  float mx = fmaxf(fmaxf(l0, l1), fmaxf(l2, l3));
  float w0 = __expf(l0 - mx), w1 = __expf(l1 - mx);
  float w2 = __expf(l2 - mx), w3 = __expf(l3 - mx);
  float inv = 1.0f / (w0 + w1 + w2 + w3);
  float o = so[(base + k0) * DH + e] * w0 + so[(base + k1) * DH + e] * w1 +
            so[(base + k2) * DH + e] * w2 + so[(base + k3) * DH + e] * w3;
  int b = bh >> 3, hd = bh & 7;
  attn[((size_t)b * L + t) * D + hd * DH + e] = o * inv;
}

// ---------------------------------------------------------------------------
// Row layernorm over D=512 (in place), one block per row.
// ---------------------------------------------------------------------------
__global__ __launch_bounds__(256) void layernorm_kernel(
    float* __restrict__ X, const float* __restrict__ g, const float* __restrict__ bta) {
  __shared__ float red[256];
  size_t row = blockIdx.x;
  int tid = threadIdx.x;
  float a = X[row * D + tid];
  float b = X[row * D + 256 + tid];
  red[tid] = a + b;
  __syncthreads();
  for (int s = 128; s > 0; s >>= 1) {
    if (tid < s) red[tid] += red[tid + s];
    __syncthreads();
  }
  float mean = red[0] * (1.0f / (float)D);
  __syncthreads();
  float da = a - mean, db = b - mean;
  red[tid] = da * da + db * db;
  __syncthreads();
  for (int s = 128; s > 0; s >>= 1) {
    if (tid < s) red[tid] += red[tid + s];
    __syncthreads();
  }
  float inv = rsqrtf(red[0] * (1.0f / (float)D) + 1e-5f);
  X[row * D + tid]       = da * inv * g[tid] + bta[tid];
  X[row * D + 256 + tid] = db * inv * g[256 + tid] + bta[256 + tid];
}

// ---------------------------------------------------------------------------
// Final projection on the last PRED tokens: (D) -> (COUT=7)
// ---------------------------------------------------------------------------
__global__ __launch_bounds__(64) void proj_kernel(
    const float* __restrict__ Hf, const float* __restrict__ Wp,
    const float* __restrict__ bp, float* __restrict__ out) {
  int idx = blockIdx.x;                 // b*PRED + tp
  int b = idx / PRED, tp = idx % PRED;
  int t = SEQ + tp;
  const float* row = Hf + ((size_t)b * L + t) * D;
  int tid = threadIdx.x;
  if (tid < COUT) {
    float acc = bp[tid];
    for (int d = 0; d < D; ++d) acc += row[d] * Wp[(size_t)d * COUT + tid];
    out[(size_t)idx * COUT + tid] = acc;
  }
}

// ---------------------------------------------------------------------------
// Host orchestration
// ---------------------------------------------------------------------------
extern "C" void kernel_launch(void* const* d_in, const int* in_sizes, int n_in,
                              void* d_out, int out_size, void* d_ws, size_t ws_size,
                              hipStream_t stream) {
  (void)in_sizes; (void)n_in; (void)out_size; (void)ws_size;
  const float* x_enc   = (const float*)d_in[0];
  const float* x_dec   = (const float*)d_in[1];
  const int*   xm_enc  = (const int*)d_in[2];
  const int*   xm_dec  = (const int*)d_in[3];
  const float* conv_w  = (const float*)d_in[4];
  const float* month_e = (const float*)d_in[5];
  const float* day_e   = (const float*)d_in[6];
  const float* wday_e  = (const float*)d_in[7];
  const float* hour_e  = (const float*)d_in[8];
  const float* min_e   = (const float*)d_in[9];
  const float* W_qk    = (const float*)d_in[10];
  const float* W_v     = (const float*)d_in[11];
  const float* W_out   = (const float*)d_in[12];
  const float* b_out   = (const float*)d_in[13];
  const float* rots    = (const float*)d_in[14];
  const float* ln1_g   = (const float*)d_in[15];
  const float* ln1_b   = (const float*)d_in[16];
  const float* W_ff1   = (const float*)d_in[17];
  const float* b_ff1   = (const float*)d_in[18];
  const float* W_ff2   = (const float*)d_in[19];
  const float* b_ff2   = (const float*)d_in[20];
  const float* ln2_g   = (const float*)d_in[21];
  const float* ln2_b   = (const float*)d_in[22];
  const float* lnf_g   = (const float*)d_in[23];
  const float* lnf_b   = (const float*)d_in[24];
  const float* W_proj  = (const float*)d_in[25];
  const float* b_proj  = (const float*)d_in[26];
  float* out = (float*)d_out;

  char* ws = (char*)d_ws;
  size_t off = 0;
  auto take = [&](size_t bytes) -> char* {
    char* p = ws + off;
    off = (off + bytes + 255) & ~(size_t)255;
    return p;
  };
  float*  P0   = (float*)take((size_t)M * D * 4);
  float*  P1   = (float*)take((size_t)M * D * 4);
  float*  P2   = (float*)take((size_t)M * D * 4);
  float*  P3   = (float*)take((size_t)M * D * 4);
  float*  BIG  = (float*)take((size_t)M * FF * 4);   // so (BH*SL*DH) == M*FF floats
  float*  slg  = (float*)take((size_t)BH * SL * 4);
  int*    stb  = (int*)take((size_t)BH * SL * 4);
  int*    undb = (int*)take((size_t)BH * SL * 4);
  int*    bck  = (int*)take((size_t)BH * NHASH * L * 4);
  __bf16* ABF  = (__bf16*)take((size_t)M * FF * 2);  // bf16 activation arena
  constexpr size_t WPL = (size_t)3 * D * D + 2 * (size_t)D * FF;
  __bf16* WT   = (__bf16*)take(WPL * EL * 2);        // bf16 transposed weights

  // Pre-convert + transpose all weights to bf16 (Wt[n][k] = W[k][n]).
  for (int l = 0; l < EL; ++l) {
    __bf16* base = WT + (size_t)l * WPL;
    wtrans_kernel<<<dim3(D / 32, D / 8), 256, 0, stream>>>(W_qk + (size_t)l * D * D, base, D, D);
    wtrans_kernel<<<dim3(D / 32, D / 8), 256, 0, stream>>>(W_v + (size_t)l * D * D, base + (size_t)D * D, D, D);
    wtrans_kernel<<<dim3(D / 32, D / 8), 256, 0, stream>>>(W_out + (size_t)l * D * D, base + (size_t)2 * D * D, D, D);
    wtrans_kernel<<<dim3(D / 32, FF / 8), 256, 0, stream>>>(W_ff1 + (size_t)l * D * FF, base + (size_t)3 * D * D, D, FF);
    wtrans_kernel<<<dim3(FF / 32, D / 8), 256, 0, stream>>>(W_ff2 + (size_t)l * FF * D, base + (size_t)3 * D * D + (size_t)D * FF, FF, D);
  }

  embed_kernel<<<M, 256, 0, stream>>>(x_enc, x_dec, xm_enc, xm_dec, conv_w,
                                      month_e, day_e, wday_e, hour_e, min_e, P0);

  float* cur = P0;
  float* a = P1;
  float* b2 = P2;
  for (int l = 0; l < EL; ++l) {
    __bf16* wqk_t  = WT + (size_t)l * WPL;
    __bf16* wv_t   = wqk_t + (size_t)D * D;
    __bf16* wout_t = wqk_t + (size_t)2 * D * D;
    __bf16* wff1_t = wqk_t + (size_t)3 * D * D;
    __bf16* wff2_t = wff1_t + (size_t)D * FF;

    // QK / V projections
    tobf16_kernel<<<(M * D) / 256, 256, 0, stream>>>(cur, ABF, (size_t)M * D);
    gemm_kernel<<<dim3(M / 128, D / 64), 256, 0, stream>>>(ABF, wqk_t, nullptr, nullptr, a, M, D, D, 0);
    gemm_kernel<<<dim3(M / 128, D / 64), 256, 0, stream>>>(ABF, wv_t, nullptr, nullptr, b2, M, D, D, 0);
    // LSH hashing + stable bucket sort
    hash_kernel<<<BH * L, 128, 0, stream>>>(a, rots + (size_t)l * DH * NHASH * (NB / 2), bck);
    sort_kernel<<<BH * NHASH, 128, 0, stream>>>(bck, stb, undb);
    // bucketed attention + hash-round combine
    attn_kernel<<<dim3(NCH, BH), 256, 0, stream>>>(a, b2, stb, BIG, slg);
    combine_kernel<<<dim3(L / 4, BH), 256, 0, stream>>>(BIG, slg, undb, P3);
    // output projection + residual, LN1
    tobf16_kernel<<<(M * D) / 256, 256, 0, stream>>>(P3, ABF, (size_t)M * D);
    gemm_kernel<<<dim3(M / 128, D / 64), 256, 0, stream>>>(ABF, wout_t, b_out + (size_t)l * D, cur, a, M, D, D, 0);
    layernorm_kernel<<<M, 256, 0, stream>>>(a, ln1_g + (size_t)l * D, ln1_b + (size_t)l * D);
    // FFN: gelu(x@W1+b1)@W2+b2 + residual, LN2
    tobf16_kernel<<<(M * D) / 256, 256, 0, stream>>>(a, ABF, (size_t)M * D);
    gemm_kernel<<<dim3(M / 128, FF / 64), 256, 0, stream>>>(ABF, wff1_t, b_ff1 + (size_t)l * FF, nullptr, BIG, M, FF, D, 1);
    tobf16_kernel<<<(M * FF) / 256, 256, 0, stream>>>(BIG, ABF, (size_t)M * FF);
    gemm_kernel<<<dim3(M / 128, D / 64), 256, 0, stream>>>(ABF, wff2_t, b_ff2 + (size_t)l * D, a, b2, M, D, FF, 0);
    layernorm_kernel<<<M, 256, 0, stream>>>(b2, ln2_g + (size_t)l * D, ln2_b + (size_t)l * D);
    cur = b2;
    if (l == 0) { a = P0; b2 = P1; }
  }
  layernorm_kernel<<<M, 256, 0, stream>>>(cur, lnf_g, lnf_b);
  proj_kernel<<<Bsz * PRED, 64, 0, stream>>>(cur, W_proj, b_proj, out);
}
